// MoELayer_66340064854168
// MI455X (gfx1250) — matrix operand order repaired and checked by reading, not compiled, not run on hardware
//
#include <hip/hip_runtime.h>
#include <hip/hip_bf16.h>
#include <math.h>

// ---------------------------------------------------------------------------
// MoE layer for gfx1250 (MI455X): bf16 WMMA grouped-GEMM, expert-gathered,
// double-buffered LDS, async global->LDS copies via inline CDNA5 asm.
// ---------------------------------------------------------------------------

typedef __attribute__((ext_vector_type(16))) __bf16 v16bf;
typedef __attribute__((ext_vector_type(8)))  float  v8f;

#define DMODEL 1024
#define DFF    4096
#define NEXP   8
#define MT     128   // M tile rows
#define NT     128   // N tile cols
#define KT     32    // K step (bf16 wmma k=32)

#if __has_builtin(__builtin_amdgcn_s_wait_asynccnt)
#define HAVE_ASYNC_LDS 1
// GLOBAL_LOAD_ASYNC_TO_LDS_B128, GV mode: vdst = per-lane LDS byte address
// (low 32 bits of the generic pointer per the flat->LDS aperture rule),
// vaddr = per-lane 64-bit global address. Tracked by ASYNCcnt.
__device__ __forceinline__ void async_copy_b128(const unsigned short* gsrc,
                                                unsigned short* ldst) {
    unsigned lds_off = (unsigned)(unsigned long long)ldst;
    asm volatile("global_load_async_to_lds_b128 %0, %1, off"
                 :
                 : "v"(lds_off), "v"((unsigned long long)gsrc)
                 : "memory");
}
__device__ __forceinline__ void async_wait0() {
    __builtin_amdgcn_s_wait_asynccnt(0);
}
#else
#define HAVE_ASYNC_LDS 0
#endif

union FragBF {
    v16bf v;
    uint4 q[2];
};

__device__ __forceinline__ unsigned short f2bf(float f) {
    unsigned int u = __float_as_uint(f);
    unsigned int r = u + 0x7FFFu + ((u >> 16) & 1u);   // round-to-nearest-even
    return (unsigned short)(r >> 16);
}

// --------------------------------------------------------------------- init
__global__ __launch_bounds__(256) void moe_init(float* __restrict__ out, int out_n,
                                                int* __restrict__ pair_tok,
                                                float* __restrict__ pair_gate, int pair_n,
                                                int* __restrict__ counts) {
    int i = blockIdx.x * 256 + threadIdx.x;
    if (i < out_n) out[i] = 0.0f;
    if (i < pair_n) { pair_tok[i] = -1; pair_gate[i] = 0.0f; }
    if (i < NEXP) counts[i] = 0;
}

// ------------------------------------------------------------- f32 -> bf16
__global__ __launch_bounds__(256) void cvt_bf16(const float* __restrict__ src,
                                                unsigned short* __restrict__ dst, int n4) {
    int i = blockIdx.x * 256 + threadIdx.x;
    if (i >= n4) return;
    float4 v = ((const float4*)src)[i];
    uint2 o;
    o.x = (unsigned int)f2bf(v.x) | ((unsigned int)f2bf(v.y) << 16);
    o.y = (unsigned int)f2bf(v.z) | ((unsigned int)f2bf(v.w) << 16);
    ((uint2*)dst)[i] = o;
}

// ------------------------------------------------------------------ router
__global__ __launch_bounds__(256) void moe_router(const float* __restrict__ x,
                                                  const float* __restrict__ rw,
                                                  const float* __restrict__ rb,
                                                  int* __restrict__ counts,
                                                  int* __restrict__ tok_e,
                                                  float* __restrict__ tok_g, int T) {
    __shared__ float s_rw[DMODEL * NEXP];          // 32 KB
    for (int i = threadIdx.x; i < DMODEL * NEXP; i += 256) s_rw[i] = rw[i];
    __syncthreads();

    int t = blockIdx.x * 256 + threadIdx.x;
    if (t >= T) return;

    const float* xr = x + (size_t)t * DMODEL;
    float lg[NEXP];
    #pragma unroll
    for (int j = 0; j < NEXP; ++j) lg[j] = rb[j];
    for (int d = 0; d < DMODEL; d += 4) {
        float4 xv = *(const float4*)(xr + d);
        #pragma unroll
        for (int j = 0; j < NEXP; ++j) {
            lg[j] += xv.x * s_rw[(d + 0) * NEXP + j] + xv.y * s_rw[(d + 1) * NEXP + j] +
                     xv.z * s_rw[(d + 2) * NEXP + j] + xv.w * s_rw[(d + 3) * NEXP + j];
        }
    }
    // top-2 (first-index wins ties, matching lax.top_k)
    int i0 = 0; float v0 = lg[0];
    #pragma unroll
    for (int j = 1; j < NEXP; ++j) if (lg[j] > v0) { v0 = lg[j]; i0 = j; }
    int i1 = -1; float v1 = -3.4e38f;
    #pragma unroll
    for (int j = 0; j < NEXP; ++j) if (j != i0 && lg[j] > v1) { v1 = lg[j]; i1 = j; }
    // softmax over the two selected logits (v0 >= v1)
    float e1 = __expf(v1 - v0);
    float inv = 1.0f / (1.0f + e1);
    tok_e[t * 2 + 0] = i0;  tok_g[t * 2 + 0] = inv;
    tok_e[t * 2 + 1] = i1;  tok_g[t * 2 + 1] = e1 * inv;
    atomicAdd(&counts[i0], 1);
    atomicAdd(&counts[i1], 1);
}

// -------------------------------------------------------------------- scan
__global__ void moe_scan(const int* __restrict__ counts, int* __restrict__ offs,
                         int* __restrict__ cursors, int* __restrict__ tile_e,
                         int* __restrict__ tile_r, int max_tiles) {
    if (threadIdx.x != 0 || blockIdx.x != 0) return;
    int off = 0, tt = 0;
    for (int e = 0; e < NEXP; ++e) {
        offs[e] = off;
        cursors[e] = 0;
        int nt = (counts[e] + MT - 1) / MT;
        for (int i = 0; i < nt; ++i) { tile_e[tt] = e; tile_r[tt] = off + i * MT; ++tt; }
        off += nt * MT;
    }
    offs[NEXP] = off;
    for (; tt < max_tiles; ++tt) { tile_e[tt] = -1; tile_r[tt] = 0; }
}

// ----------------------------------------------------------------- scatter
__global__ __launch_bounds__(256) void moe_scatter(const int* __restrict__ tok_e,
                                                   const float* __restrict__ tok_g,
                                                   const int* __restrict__ offs,
                                                   int* __restrict__ cursors,
                                                   int* __restrict__ pair_tok,
                                                   float* __restrict__ pair_gate, int T) {
    int t = blockIdx.x * 256 + threadIdx.x;
    if (t >= T) return;
    #pragma unroll
    for (int k = 0; k < 2; ++k) {
        int e = tok_e[t * 2 + k];
        int pos = atomicAdd(&cursors[e], 1);
        int idx = offs[e] + pos;
        pair_tok[idx] = t;
        pair_gate[idx] = tok_g[t * 2 + k];
    }
}

// --------------------------------------------------------- GEMM1: x@w1 + gelu
__global__ __launch_bounds__(256) void moe_gemm1(const unsigned short* __restrict__ x16,
                                                 const unsigned short* __restrict__ w1_16,
                                                 const float* __restrict__ b1,
                                                 const int* __restrict__ tile_e,
                                                 const int* __restrict__ tile_r,
                                                 const int* __restrict__ pair_tok,
                                                 unsigned short* __restrict__ h16) {
    int e = tile_e[blockIdx.x];
    if (e < 0) return;                       // padded tile, uniform exit
    int row0 = tile_r[blockIdx.x];
    int n0 = blockIdx.y * NT;                // within DFF

    __shared__ unsigned short As[2][MT][KT]; // 2 x 8 KB, [m][k]
    __shared__ unsigned short Bs[2][NT][KT]; // 2 x 8 KB, [n][k] (transposed stage)

    const int tid  = threadIdx.x;
    const int lane = tid & 31;
    const int wave = tid >> 5;
    const int wm   = wave >> 2;              // 0..1 -> 64-row strip
    const int wn   = wave & 3;               // 0..3 -> 32-col strip
    const int l15  = lane & 15;
    const int hi   = lane >> 4;

    v8f acc[4][2];
    #pragma unroll
    for (int mi = 0; mi < 4; ++mi)
        #pragma unroll
        for (int ni = 0; ni < 2; ++ni) acc[mi][ni] = (v8f)0.0f;

    const int arow  = tid >> 1;              // 0..127
    const int ahalf = (tid & 1) * 16;        // ushort offset 0 / 16
    const long long atok = pair_tok[row0 + arow];
    const unsigned short* asrc = (atok >= 0) ? (x16 + (size_t)atok * DMODEL + ahalf) : nullptr;
    const int bk  = tid >> 3;                // 0..31
    const int bnc = (tid & 7) * 16;          // 0..112
    const unsigned short* wbase = w1_16 + (size_t)e * DMODEL * DFF + n0 + bnc;

    // pre-zero padded gather rows once (both buffers); never rewritten
    if (atok < 0) {
        uint4 z = {0, 0, 0, 0};
        #pragma unroll
        for (int b = 0; b < 2; ++b) {
            uint4* d = (uint4*)&As[b][arow][ahalf];
            d[0] = z; d[1] = z;
        }
    }

    // ---- prologue: stage tile kb=0 into buffer 0
    {
        if (atok >= 0) {
#if HAVE_ASYNC_LDS
            async_copy_b128(asrc, &As[0][arow][ahalf]);
            async_copy_b128(asrc + 8, &As[0][arow][ahalf + 8]);
#else
            uint4* d = (uint4*)&As[0][arow][ahalf];
            d[0] = ((const uint4*)asrc)[0];
            d[1] = ((const uint4*)asrc)[1];
#endif
        }
        const unsigned short* src = wbase + (size_t)bk * DFF;
        unsigned short tmp[16];
        *(uint4*)&tmp[0] = ((const uint4*)src)[0];
        *(uint4*)&tmp[8] = ((const uint4*)src)[1];
        #pragma unroll
        for (int i = 0; i < 16; ++i) Bs[0][bnc + i][bk] = tmp[i];
    }
#if HAVE_ASYNC_LDS
    async_wait0();
#endif
    __syncthreads();

    for (int kb = 0; kb < DMODEL; kb += KT) {
        const int cur = (kb / KT) & 1;
        const int nxt = cur ^ 1;
        const bool has_next = (kb + KT) < DMODEL;

        // 1) kick off next A tile into the other buffer
#if HAVE_ASYNC_LDS
        if (has_next && atok >= 0) {
            const unsigned short* g = asrc + kb + KT;
            async_copy_b128(g, &As[nxt][arow][ahalf]);
            async_copy_b128(g + 8, &As[nxt][arow][ahalf + 8]);
        }
#else
        uint4 a0 = {0, 0, 0, 0}, a1 = {0, 0, 0, 0};
        if (has_next && atok >= 0) {
            const uint4* g = (const uint4*)(asrc + kb + KT);
            a0 = g[0]; a1 = g[1];
        }
#endif
        // 2) next B tile into registers
        unsigned short btmp[16];
        if (has_next) {
            const unsigned short* src = wbase + (size_t)(kb + KT + bk) * DFF;
            *(uint4*)&btmp[0] = ((const uint4*)src)[0];
            *(uint4*)&btmp[8] = ((const uint4*)src)[1];
        }

        // 3) compute on current buffer: 8 wmma per wave
        FragBF a[4], b[2];
        #pragma unroll
        for (int mi = 0; mi < 4; ++mi) {
            const unsigned short* p = &As[cur][wm * 64 + mi * 16 + l15][hi * 8];
            a[mi].q[0] = *(const uint4*)p;          // K {0..7} / {8..15}
            a[mi].q[1] = *(const uint4*)(p + 16);   // K {16..23} / {24..31}
        }
        #pragma unroll
        for (int ni = 0; ni < 2; ++ni) {
            const unsigned short* p = &Bs[cur][wn * 32 + ni * 16 + l15][hi * 8];
            b[ni].q[0] = *(const uint4*)p;
            b[ni].q[1] = *(const uint4*)(p + 16);
        }
        #pragma unroll
        for (int mi = 0; mi < 4; ++mi)
            #pragma unroll
            for (int ni = 0; ni < 2; ++ni)
                acc[mi][ni] = __builtin_amdgcn_wmma_f32_16x16x32_bf16(
                    false, a[mi].v, false, b[ni].v, (short)0, acc[mi][ni], false, false);

        // 4) publish next tiles
        if (has_next) {
#if !HAVE_ASYNC_LDS
            if (atok >= 0) {
                uint4* d = (uint4*)&As[nxt][arow][ahalf];
                d[0] = a0; d[1] = a1;
            }
#endif
            #pragma unroll
            for (int i = 0; i < 16; ++i) Bs[nxt][bnc + i][bk] = btmp[i];
        }
#if HAVE_ASYNC_LDS
        async_wait0();
#endif
        __syncthreads();
    }

    // ---- epilogue: exact GELU, store bf16 h
    const float* b1e = b1 + (size_t)e * DFF;
    #pragma unroll
    for (int mi = 0; mi < 4; ++mi)
        #pragma unroll
        for (int ni = 0; ni < 2; ++ni) {
            int fcol = n0 + wn * 32 + ni * 16 + l15;
            float bias = b1e[fcol];
            #pragma unroll
            for (int v = 0; v < 8; ++v) {
                int m = wm * 64 + mi * 16 + hi * 8 + v;
                float val = acc[mi][ni][v] + bias;
                val = 0.5f * val * (1.0f + erff(val * 0.70710678118654752f));
                h16[(size_t)(row0 + m) * DFF + fcol] = f2bf(val);
            }
        }
}

// ----------------------------------------------- GEMM2: h@w2, gated scatter-add
__global__ __launch_bounds__(256) void moe_gemm2(const unsigned short* __restrict__ h16,
                                                 const unsigned short* __restrict__ w2_16,
                                                 const float* __restrict__ b2,
                                                 const float* __restrict__ pair_gate,
                                                 const int* __restrict__ tile_e,
                                                 const int* __restrict__ tile_r,
                                                 const int* __restrict__ pair_tok,
                                                 float* __restrict__ out) {
    int e = tile_e[blockIdx.x];
    if (e < 0) return;
    int row0 = tile_r[blockIdx.x];
    int n0 = blockIdx.y * NT;                // within DMODEL

    __shared__ unsigned short As[2][MT][KT];
    __shared__ unsigned short Bs[2][NT][KT];

    const int tid  = threadIdx.x;
    const int lane = tid & 31;
    const int wave = tid >> 5;
    const int wm   = wave >> 2;
    const int wn   = wave & 3;
    const int l15  = lane & 15;
    const int hi   = lane >> 4;

    v8f acc[4][2];
    #pragma unroll
    for (int mi = 0; mi < 4; ++mi)
        #pragma unroll
        for (int ni = 0; ni < 2; ++ni) acc[mi][ni] = (v8f)0.0f;

    const int arow  = tid >> 1;
    const int ahalf = (tid & 1) * 16;
    const unsigned short* asrc = h16 + (size_t)(row0 + arow) * DFF + ahalf;
    const int bk  = tid >> 3;
    const int bnc = (tid & 7) * 16;
    const unsigned short* wbase = w2_16 + (size_t)e * DFF * DMODEL + n0 + bnc;

    // ---- prologue: stage tile kb=0 into buffer 0
    {
#if HAVE_ASYNC_LDS
        async_copy_b128(asrc, &As[0][arow][ahalf]);
        async_copy_b128(asrc + 8, &As[0][arow][ahalf + 8]);
#else
        uint4* d = (uint4*)&As[0][arow][ahalf];
        d[0] = ((const uint4*)asrc)[0];
        d[1] = ((const uint4*)asrc)[1];
#endif
        const unsigned short* src = wbase + (size_t)bk * DMODEL;
        unsigned short tmp[16];
        *(uint4*)&tmp[0] = ((const uint4*)src)[0];
        *(uint4*)&tmp[8] = ((const uint4*)src)[1];
        #pragma unroll
        for (int i = 0; i < 16; ++i) Bs[0][bnc + i][bk] = tmp[i];
    }
#if HAVE_ASYNC_LDS
    async_wait0();
#endif
    __syncthreads();

    for (int kb = 0; kb < DFF; kb += KT) {
        const int cur = (kb / KT) & 1;
        const int nxt = cur ^ 1;
        const bool has_next = (kb + KT) < DFF;

#if HAVE_ASYNC_LDS
        if (has_next) {
            const unsigned short* g = asrc + kb + KT;
            async_copy_b128(g, &As[nxt][arow][ahalf]);
            async_copy_b128(g + 8, &As[nxt][arow][ahalf + 8]);
        }
#else
        uint4 a0, a1;
        if (has_next) {
            const uint4* g = (const uint4*)(asrc + kb + KT);
            a0 = g[0]; a1 = g[1];
        }
#endif
        unsigned short btmp[16];
        if (has_next) {
            const unsigned short* src = wbase + (size_t)(kb + KT + bk) * DMODEL;
            *(uint4*)&btmp[0] = ((const uint4*)src)[0];
            *(uint4*)&btmp[8] = ((const uint4*)src)[1];
        }

        FragBF a[4], b[2];
        #pragma unroll
        for (int mi = 0; mi < 4; ++mi) {
            const unsigned short* p = &As[cur][wm * 64 + mi * 16 + l15][hi * 8];
            a[mi].q[0] = *(const uint4*)p;
            a[mi].q[1] = *(const uint4*)(p + 16);
        }
        #pragma unroll
        for (int ni = 0; ni < 2; ++ni) {
            const unsigned short* p = &Bs[cur][wn * 32 + ni * 16 + l15][hi * 8];
            b[ni].q[0] = *(const uint4*)p;
            b[ni].q[1] = *(const uint4*)(p + 16);
        }
        #pragma unroll
        for (int mi = 0; mi < 4; ++mi)
            #pragma unroll
            for (int ni = 0; ni < 2; ++ni)
                acc[mi][ni] = __builtin_amdgcn_wmma_f32_16x16x32_bf16(
                    false, a[mi].v, false, b[ni].v, (short)0, acc[mi][ni], false, false);

        if (has_next) {
#if !HAVE_ASYNC_LDS
            uint4* d = (uint4*)&As[nxt][arow][ahalf];
            d[0] = a0; d[1] = a1;
#endif
            #pragma unroll
            for (int i = 0; i < 16; ++i) Bs[nxt][bnc + i][bk] = btmp[i];
        }
#if HAVE_ASYNC_LDS
        async_wait0();
#endif
        __syncthreads();
    }

    // ---- epilogue: (acc + b2) * gate scatter-add (token appears in 2 experts)
    const float* b2e = b2 + (size_t)e * DMODEL;
    #pragma unroll
    for (int mi = 0; mi < 4; ++mi)
        #pragma unroll
        for (int ni = 0; ni < 2; ++ni) {
            int dcol = n0 + wn * 32 + ni * 16 + l15;
            float bias = b2e[dcol];
            #pragma unroll
            for (int v = 0; v < 8; ++v) {
                int m = wm * 64 + mi * 16 + hi * 8 + v;
                int prow = row0 + m;
                int tok = pair_tok[prow];
                if (tok >= 0) {
                    float g = pair_gate[prow];
                    atomicAdd(&out[(size_t)tok * DMODEL + dcol], (acc[mi][ni][v] + bias) * g);
                }
            }
        }
}

// ---------------------------------------------------------------------------
extern "C" void kernel_launch(void* const* d_in, const int* in_sizes, int n_in,
                              void* d_out, int out_size, void* d_ws, size_t ws_size,
                              hipStream_t stream) {
    const float* x  = (const float*)d_in[0];
    const float* rw = (const float*)d_in[1];
    const float* rb = (const float*)d_in[2];
    const float* w1 = (const float*)d_in[3];
    const float* b1 = (const float*)d_in[4];
    const float* w2 = (const float*)d_in[5];
    const float* b2 = (const float*)d_in[6];
    float* out = (float*)d_out;

    const int T = in_sizes[0] / DMODEL;           // B*S tokens
    const int PAIR_CAP  = 2 * T + NEXP * MT;      // routed pairs + per-expert pad
    const int MAX_TILES = PAIR_CAP / MT;

    // ---- workspace carve-out (256 B aligned)
    char* w = (char*)d_ws;
    size_t used = 0;
    auto carve = [&](size_t bytes) {
        char* p = w;
        size_t pad = (bytes + 255) & ~(size_t)255;
        w += pad; used += pad;
        return p;
    };
    unsigned short* x16   = (unsigned short*)carve((size_t)T * DMODEL * 2);
    unsigned short* w1_16 = (unsigned short*)carve((size_t)NEXP * DMODEL * DFF * 2);
    unsigned short* w2_16 = (unsigned short*)carve((size_t)NEXP * DFF * DMODEL * 2);
    unsigned short* h16   = (unsigned short*)carve((size_t)PAIR_CAP * DFF * 2);
    int*   pair_tok  = (int*)carve((size_t)PAIR_CAP * 4);
    float* pair_gate = (float*)carve((size_t)PAIR_CAP * 4);
    int*   tok_e     = (int*)carve((size_t)T * 2 * 4);
    float* tok_g     = (float*)carve((size_t)T * 2 * 4);
    int*   counts    = (int*)carve(64);
    int*   cursors   = (int*)carve(64);
    int*   offs      = (int*)carve(64);
    int*   tile_e    = (int*)carve((size_t)MAX_TILES * 4);
    int*   tile_r    = (int*)carve((size_t)MAX_TILES * 4);
    if (used > ws_size) return;  // workspace too small: deterministic no-op

    const int out_n = T * DMODEL;
    const int initN = out_n > PAIR_CAP ? out_n : PAIR_CAP;

    moe_init<<<(initN + 255) / 256, 256, 0, stream>>>(out, out_n, pair_tok, pair_gate,
                                                      PAIR_CAP, counts);
    cvt_bf16<<<((T * DMODEL / 4) + 255) / 256, 256, 0, stream>>>(x, x16, T * DMODEL / 4);
    cvt_bf16<<<((NEXP * DMODEL * DFF / 4) + 255) / 256, 256, 0, stream>>>(
        w1, w1_16, NEXP * DMODEL * DFF / 4);
    cvt_bf16<<<((NEXP * DFF * DMODEL / 4) + 255) / 256, 256, 0, stream>>>(
        w2, w2_16, NEXP * DFF * DMODEL / 4);
    moe_router<<<(T + 255) / 256, 256, 0, stream>>>(x, rw, rb, counts, tok_e, tok_g, T);
    moe_scan<<<1, 64, 0, stream>>>(counts, offs, cursors, tile_e, tile_r, MAX_TILES);
    moe_scatter<<<(T + 255) / 256, 256, 0, stream>>>(tok_e, tok_g, offs, cursors,
                                                     pair_tok, pair_gate, T);
    moe_gemm1<<<dim3(MAX_TILES, DFF / NT), 256, 0, stream>>>(x16, w1_16, b1, tile_e,
                                                             tile_r, pair_tok, h16);
    moe_gemm2<<<dim3(MAX_TILES, DMODEL / NT), 256, 0, stream>>>(h16, w2_16, b2, pair_gate,
                                                                tile_e, tile_r, pair_tok, out);
}
